// MoE_63015760167571
// MI455X (gfx1250) — compile-verified, matrix operand order checked by reading, not run on hardware
//
#include <hip/hip_runtime.h>
#include <hip/hip_bf16.h>
#include <math.h>

#define D_MODEL 1024
#define H_DIM   4096
#define S_TOK   2048
#define N_EXP   8
#define TOP_K   2
#define TILE_M  32      // tokens per block (2 WMMA m-subtiles)
#define NWAVE   8       // 256 threads
#define BIGC    256     // h columns per outer iteration (8 waves x 32)

typedef __attribute__((ext_vector_type(16))) __bf16 v16bf;
typedef __attribute__((ext_vector_type(8)))  float  v8f;

__device__ __forceinline__ __bf16 f2bf(float f) { return (__bf16)f; }

__device__ __forceinline__ v8f wmma_bf16(v16bf a, v16bf b, v8f c) {
    // 8 args: (neg_a, A, neg_b, B, c_mod, C, reuse_a, reuse_b)
    return __builtin_amdgcn_wmma_f32_16x16x32_bf16(false, a, false, b, (short)0, c,
                                                   false, false);
}

// WMMA 16x32 bf16 A/B fragment element position (ISA 7.12.2):
// lanes 0-15 hold K={0..7,16..23}, lanes 16-31 hold K={8..15,24..31}.
__device__ __forceinline__ int frag_half(int kk) { return (kk & 8) ? 1 : 0; }
__device__ __forceinline__ int frag_elem(int kk) { return (kk & 7) + ((kk & 16) ? 8 : 0); }

// B fragment (32x16, KxN) gathered from global fp32 with row stride `stride`;
// base points at element [k=0][n=0] of the tile; converted to bf16 on the fly.
__device__ __forceinline__ v16bf load_b_f32(const float* __restrict__ base,
                                            int stride, int lane) {
    int n = lane & 15, half = lane >> 4;
    v16bf b;
#pragma unroll
    for (int r = 0; r < 8; ++r) {
        int kb = (r < 4 ? 2 * r : 16 + 2 * (r - 4)) + half * 8;
        b[2 * r]     = f2bf(base[(size_t)kb * stride + n]);
        b[2 * r + 1] = f2bf(base[(size_t)(kb + 1) * stride + n]);
    }
    return b;
}

__device__ __forceinline__ float fast_silu_mul(float g, float u) {
    float sig = __builtin_amdgcn_rcpf(1.f + __expf(-g));  // v_rcp_f32
    return g * sig * u;
}

// ---------------------------------------------------------------------------
// Weight packer: W[Ktot, N] fp32 row-major -> fragment-ordered bf16 tiles.
// Tile (kc, nt) = rows kc*32..+31, cols nt*16..+15; stored as 32 lanes x 32 B.
// Tile order: kc-major, then nt. One wave packs one tile.
// ---------------------------------------------------------------------------
__global__ __launch_bounds__(256) void moe_pack_kernel(
    const float* __restrict__ W, v16bf* __restrict__ P, int Ktot, int N) {
    int tileId = blockIdx.x * NWAVE + (int)(threadIdx.x >> 5);
    int lane   = threadIdx.x & 31;
    int ntn    = N >> 4;
    int kc     = tileId / ntn;
    int nt     = tileId - kc * ntn;
    if (kc >= (Ktot >> 5)) return;
    v16bf b = load_b_f32(W + (size_t)kc * 32 * N + nt * 16, N, lane);
    P[(size_t)tileId * 32 + lane] = b;  // 2x global_store_b128 per lane
}

// ---------------------------------------------------------------------------
// Gate: logits -> softmax -> top-2 -> renormalized combine[S,E]; aux partials.
// ---------------------------------------------------------------------------
__global__ __launch_bounds__(256) void moe_gate_kernel(
    const float* __restrict__ x, const float* __restrict__ wg,
    float* __restrict__ combine, float* __restrict__ sums) {
    __shared__ float lwg[D_MODEL * N_EXP];  // 32 KB
    int tid = threadIdx.x;
    for (int i = tid; i < D_MODEL * N_EXP; i += 256) lwg[i] = wg[i];
    __syncthreads();

    int s = blockIdx.x * 256 + tid;
    float acc[N_EXP];
#pragma unroll
    for (int e = 0; e < N_EXP; ++e) acc[e] = 0.f;
    const float* xr = x + (size_t)s * D_MODEL;
    for (int k = 0; k < D_MODEL; ++k) {
        float xv = xr[k];
#pragma unroll
        for (int e = 0; e < N_EXP; ++e) acc[e] = fmaf(xv, lwg[k * N_EXP + e], acc[e]);
    }
    float mx = acc[0];
#pragma unroll
    for (int e = 1; e < N_EXP; ++e) mx = fmaxf(mx, acc[e]);
    float p[N_EXP], den = 0.f;
#pragma unroll
    for (int e = 0; e < N_EXP; ++e) { p[e] = __expf(acc[e] - mx); den += p[e]; }
    float inv = __builtin_amdgcn_rcpf(den);
#pragma unroll
    for (int e = 0; e < N_EXP; ++e) p[e] *= inv;
    int i0 = 0; float v0 = p[0];
#pragma unroll
    for (int e = 1; e < N_EXP; ++e) if (p[e] > v0) { v0 = p[e]; i0 = e; }
    int i1 = (i0 == 0) ? 1 : 0; float v1 = p[i1];
#pragma unroll
    for (int e = 0; e < N_EXP; ++e)
        if (e != i0 && p[e] > v1) { v1 = p[e]; i1 = e; }
    float rn = __builtin_amdgcn_rcpf(v0 + v1);
    float w0 = v0 * rn, w1 = v1 * rn;
#pragma unroll
    for (int e = 0; e < N_EXP; ++e) {
        float c = (e == i0) ? w0 : ((e == i1) ? w1 : 0.f);
        combine[(size_t)s * N_EXP + e] = c;
        unsafeAtomicAdd(&sums[8 + e], p[e]);
    }
    unsafeAtomicAdd(&sums[i0], 1.f);
    unsafeAtomicAdd(&sums[i1], 1.f);
}

// ---------------------------------------------------------------------------
// Fused expert kernel. PACKED=true: B fragments are contiguous bf16 tiles in
// workspace (2x global_load_b128/lane). PACKED=false: gather+convert fp32.
// ---------------------------------------------------------------------------
template <bool PACKED>
__global__ __launch_bounds__(256) void moe_expert_kernel(
    const float* __restrict__ x, const float* __restrict__ w1,
    const float* __restrict__ w3, const float* __restrict__ w2,
    const v16bf* __restrict__ pw1, const v16bf* __restrict__ pw3,
    const v16bf* __restrict__ pw2,
    const float* __restrict__ combine, float* __restrict__ y) {
    __shared__ v16bf lds_x[D_MODEL / 32][2][32];  // 64 KB, A-fragment order
    __shared__ v16bf lds_h[NWAVE][2][32];         // 16 KB, A-fragment order
    __shared__ float lds_c[TILE_M];

    const int tile = blockIdx.x;
    const int e    = blockIdx.y;
    const int tid  = threadIdx.x;
    const int lane = tid & 31;
    const int wave = tid >> 5;
    const int tok0 = tile * TILE_M;

    if (tid < TILE_M) lds_c[tid] = combine[(size_t)(tok0 + tid) * N_EXP + e];
    __syncthreads();

    bool any = false;
#pragma unroll
    for (int i = 0; i < TILE_M; ++i) any |= (lds_c[i] != 0.f);
    if (!any) return;  // uniform skip of unused (expert, tile) pairs

    for (int i = tid; i < TILE_M * D_MODEL; i += 256) {
        int m = i >> 10, k = i & (D_MODEL - 1);
        __bf16 v = f2bf(x[(size_t)(tok0 + m) * D_MODEL + k]);
        int kc = k >> 5, kk = k & 31;
        int ln = (m & 15) | (frag_half(kk) << 4);
        ((__bf16*)&lds_x[kc][m >> 4][ln])[frag_elem(kk)] = v;
    }
    __syncthreads();

    const float* W1 = w1 + (size_t)e * D_MODEL * H_DIM;
    const float* W3 = w3 + (size_t)e * D_MODEL * H_DIM;
    const float* W2 = w2 + (size_t)e * H_DIM * D_MODEL;
    const v16bf* P1 = pw1 + (size_t)e * (D_MODEL / 32) * (H_DIM / 16) * 32;
    const v16bf* P3 = pw3 + (size_t)e * (D_MODEL / 32) * (H_DIM / 16) * 32;
    const v16bf* P2 = pw2 + (size_t)e * (H_DIM / 32) * (D_MODEL / 16) * 32;

    v8f zero = {0.f, 0.f, 0.f, 0.f, 0.f, 0.f, 0.f, 0.f};
    v8f dacc[2][8];
#pragma unroll
    for (int mt = 0; mt < 2; ++mt)
#pragma unroll
        for (int t = 0; t < 8; ++t) dacc[mt][t] = zero;

    const int m0 = (lane >> 4) ? 8 : 0;
    const int nn = lane & 15;

    for (int hc = 0; hc < H_DIM; hc += BIGC) {
        const int colb = hc + wave * 32;
        v8f u1[2][2], u3[2][2];
#pragma unroll
        for (int mt = 0; mt < 2; ++mt)
#pragma unroll
            for (int nt = 0; nt < 2; ++nt) { u1[mt][nt] = zero; u3[mt][nt] = zero; }

        for (int kc = 0; kc < D_MODEL; kc += 32) {
            v16bf a0 = lds_x[kc >> 5][0][lane];
            v16bf a1 = lds_x[kc >> 5][1][lane];
#pragma unroll
            for (int nt = 0; nt < 2; ++nt) {
                v16bf b1, b3;
                if (PACKED) {
                    size_t tIdx = (size_t)(kc >> 5) * (H_DIM / 16) + ((colb >> 4) + nt);
                    b1 = P1[tIdx * 32 + lane];
                    b3 = P3[tIdx * 32 + lane];
                } else {
                    b1 = load_b_f32(W1 + (size_t)kc * H_DIM + colb + nt * 16, H_DIM, lane);
                    b3 = load_b_f32(W3 + (size_t)kc * H_DIM + colb + nt * 16, H_DIM, lane);
                }
                u1[0][nt] = wmma_bf16(a0, b1, u1[0][nt]);
                u1[1][nt] = wmma_bf16(a1, b1, u1[1][nt]);
                u3[0][nt] = wmma_bf16(a0, b3, u3[0][nt]);
                u3[1][nt] = wmma_bf16(a1, b3, u3[1][nt]);
            }
        }
#pragma unroll
        for (int mt = 0; mt < 2; ++mt)
#pragma unroll
            for (int nt = 0; nt < 2; ++nt)
#pragma unroll
                for (int v = 0; v < 8; ++v) {
                    int m = v + m0 + mt * 16;
                    int kk = nt * 16 + nn;
                    float hv = fast_silu_mul(u1[mt][nt][v], u3[mt][nt][v]);
                    int ln = (m & 15) | (frag_half(kk) << 4);
                    ((__bf16*)&lds_h[wave][m >> 4][ln])[frag_elem(kk)] = f2bf(hv);
                }
        __syncthreads();

        for (int s = 0; s < NWAVE; ++s) {
            v16bf a0 = lds_h[s][0][lane];
            v16bf a1 = lds_h[s][1][lane];
#pragma unroll
            for (int t = 0; t < 8; ++t) {
                v16bf b;
                if (PACKED) {
                    size_t tIdx = (size_t)((hc + s * 32) >> 5) * (D_MODEL / 16) +
                                  (wave * 8 + t);
                    b = P2[tIdx * 32 + lane];
                } else {
                    b = load_b_f32(W2 + (size_t)(hc + s * 32) * D_MODEL +
                                       wave * 128 + t * 16, D_MODEL, lane);
                }
                dacc[0][t] = wmma_bf16(a0, b, dacc[0][t]);
                dacc[1][t] = wmma_bf16(a1, b, dacc[1][t]);
            }
        }
        __syncthreads();
    }

#pragma unroll
    for (int mt = 0; mt < 2; ++mt)
#pragma unroll
        for (int t = 0; t < 8; ++t)
#pragma unroll
            for (int v = 0; v < 8; ++v) {
                int m = v + m0 + mt * 16;
                float val = dacc[mt][t][v] * lds_c[m];
                unsafeAtomicAdd(
                    &y[(size_t)(tok0 + m) * D_MODEL + wave * 128 + t * 16 + nn], val);
            }
}

__global__ void moe_aux_kernel(const float* __restrict__ sums, float* __restrict__ out_aux) {
    if (threadIdx.x == 0) {
        float a = 0.f;
#pragma unroll
        for (int e = 0; e < N_EXP; ++e) {
            float frac = sums[e] / (float)(S_TOK * TOP_K);
            float imp  = sums[8 + e] / (float)S_TOK;
            a = fmaf(frac, imp, a);
        }
        *out_aux = (float)N_EXP * a;
    }
}

extern "C" void kernel_launch(void* const* d_in, const int* in_sizes, int n_in,
                              void* d_out, int out_size, void* d_ws, size_t ws_size,
                              hipStream_t stream) {
    const float* x  = (const float*)d_in[0];
    const float* wg = (const float*)d_in[1];
    const float* w1 = (const float*)d_in[2];
    const float* w3 = (const float*)d_in[3];
    const float* w2 = (const float*)d_in[4];
    float* y = (float*)d_out;

    float* sums    = (float*)d_ws;       // 16 floats @ +0
    float* combine = (float*)d_ws + 64;  // [S, E] floats @ +256 B

    // Packed-weight region: starts at +1 MB; 3 x 64 MB of bf16 fragment tiles.
    const size_t packOff   = (size_t)1 << 20;
    const size_t tilesUp   = (size_t)(N_EXP * D_MODEL / 32) * (H_DIM / 16);   // 65536
    const size_t tilesDown = (size_t)(N_EXP * H_DIM / 32) * (D_MODEL / 16);   // 65536
    const size_t packBytes = (tilesUp * 2 + tilesDown) * 32 * sizeof(v16bf);  // ~201 MB
    const bool packed = ws_size >= packOff + packBytes;

    v16bf* pw1 = (v16bf*)((char*)d_ws + packOff);
    v16bf* pw3 = pw1 + tilesUp * 32;
    v16bf* pw2 = pw3 + tilesUp * 32;

    hipMemsetAsync(d_out, 0, (size_t)out_size * sizeof(float), stream);
    hipMemsetAsync(d_ws, 0, 16 * sizeof(float), stream);

    moe_gate_kernel<<<S_TOK / 256, 256, 0, stream>>>(x, wg, combine, sums);

    dim3 grid(S_TOK / TILE_M, N_EXP);
    if (packed) {
        moe_pack_kernel<<<(int)(tilesUp / NWAVE), 256, 0, stream>>>(
            w1, pw1, N_EXP * D_MODEL, H_DIM);
        moe_pack_kernel<<<(int)(tilesUp / NWAVE), 256, 0, stream>>>(
            w3, pw3, N_EXP * D_MODEL, H_DIM);
        moe_pack_kernel<<<(int)(tilesDown / NWAVE), 256, 0, stream>>>(
            w2, pw2, N_EXP * H_DIM, D_MODEL);
        moe_expert_kernel<true><<<grid, 256, 0, stream>>>(
            x, w1, w3, w2, pw1, pw3, pw2, combine, y);
    } else {
        moe_expert_kernel<false><<<grid, 256, 0, stream>>>(
            x, w1, w3, w2, pw1, pw3, pw2, combine, y);
    }

    moe_aux_kernel<<<1, 32, 0, stream>>>(sums, y + (size_t)S_TOK * D_MODEL);
}
